// Encoder_86148454023320
// MI455X (gfx1250) — matrix operand order repaired and checked by reading, not compiled, not run on hardware
//
#include <hip/hip_runtime.h>
#include <hip/hip_bf16.h>

// ---------------------------------------------------------------------------
// Transformer encoder forward for MI455X (gfx1250, wave32, WMMA).
// GEMMs: v_wmma_f32_16x16x32_f16 with the weight tile double-buffered in LDS
// via global_load_async_to_lds_b128 (ASYNCcnt). Attention: flash-style VALU.
// ---------------------------------------------------------------------------

typedef __attribute__((ext_vector_type(16))) _Float16 v16h;
typedef __attribute__((ext_vector_type(8)))  float    v8f;

#define B_DIM   2
#define S_DIM   2048
#define F_IN    438
#define F_PAD   448      // 438 padded up to a multiple of 32 for the K loop
#define H_DIM   768
#define NH_DIM  12
#define HD_DIM  64
#define W_WIN   32
#define L_DIM   2
#define MROWS   (B_DIM * S_DIM)   // 4096 flattened rows

#define EPI_RELU 1
#define EPI_POS  2

union FragU { v16h h; unsigned int u[8]; };

__device__ __forceinline__ unsigned short f2h(float f) {
    _Float16 h = (_Float16)f;
    return __builtin_bit_cast(unsigned short, h);
}

// CDNA5 async copy: per-lane 16B global -> LDS, tracked by ASYNCcnt.
__device__ __forceinline__ void async_copy_b128(unsigned lds_byte_addr, const void* g) {
    asm volatile("global_load_async_to_lds_b128 %0, %1, off"
                 :: "v"(lds_byte_addr), "v"((unsigned long long)(uintptr_t)g)
                 : "memory");
}
__device__ __forceinline__ void wait_asynccnt0() {
    asm volatile("s_wait_asynccnt 0x0" ::: "memory");
}

// ---------------------------------------------------------------------------
// f32 -> f16 conversion kernels (weights / padded activations)
// ---------------------------------------------------------------------------
__global__ void cvt_f32_to_f16_kernel(const float* __restrict__ in,
                                      unsigned short* __restrict__ out, int n) {
    int i = blockIdx.x * blockDim.x + threadIdx.x;
    if (i < n) out[i] = f2h(in[i]);
}

// Convert rows x kin f32 into rows x kout f16 with zero padding (kout >= kin).
__global__ void cvt_pad_kernel(const float* __restrict__ in,
                               unsigned short* __restrict__ out,
                               int rows, int kin, int kout) {
    int i = blockIdx.x * blockDim.x + threadIdx.x;
    if (i >= rows * kout) return;
    int r = i / kout, c = i - r * kout;
    out[i] = (c < kin) ? f2h(in[(size_t)r * kin + c]) : (unsigned short)0;
}

// ---------------------------------------------------------------------------
// WMMA GEMM: out[M,N] = A[M,K](f16) * W[N,K]^T(f16) + bias[N]  (f32 accum)
// Block = 256 threads = 8 waves; block tile 128x64; wave tile 32x32
// (2x2 of 16x16 WMMA tiles -> 4 v_wmma per 32-wide K step).
// Weight tile (64 rows x 32 K = 4KB) is async-copied into LDS, double
// buffered: the async engine streams tile i+1 while WMMA consumes tile i.
// A fragments are direct global b128 loads (pipelined against LOADcnt).
// ---------------------------------------------------------------------------
__global__ __launch_bounds__(256) void gemm_wmma_kernel(
    const unsigned short* __restrict__ A,     // M x K  f16
    const unsigned short* __restrict__ W,     // N x K  f16 (row-major, used transposed)
    const float* __restrict__ bias,           // N
    float* __restrict__ outF32,               // M x N or nullptr
    unsigned short* __restrict__ outF16,      // M x N or nullptr
    int M, int N, int K, int flags,
    const float* __restrict__ pos_table,      // (S+1, H) for EPI_POS
    const int* __restrict__ src_pos)          // (B*S)    for EPI_POS
{
    // Two 64x32-f16 weight tiles (4KB each), 16B-aligned for b128 traffic.
    __shared__ __align__(16) unsigned short smemB[2][64 * 32];

    const int lane = threadIdx.x & 31;
    const int wave = threadIdx.x >> 5;
    const int wm = wave & 3;          // 4 waves along M (4 * 32 = 128 rows)
    const int wn = wave >> 2;         // 2 waves along N (2 * 32 = 64 cols)
    const int r0 = blockIdx.y * 128 + wm * 32;
    const int cblk = blockIdx.x * 64;
    const int c0 = cblk + wn * 32;
    const int mL = lane & 15;
    const int hL = lane >> 4;

    // Cooperative staging: thread t moves one b128 (row = t/4, 16B chunk = t%4).
    const int trow = threadIdx.x >> 2;
    const int tch  = threadIdx.x & 3;
    const unsigned short* gsrc = W + (size_t)(cblk + trow) * K + tch * 8;
    const unsigned lds_dst0 = (unsigned)(uintptr_t)&smemB[0][trow * 32 + tch * 8];
    const unsigned lds_dst1 = (unsigned)(uintptr_t)&smemB[1][trow * 32 + tch * 8];

    v8f acc00 = {}, acc01 = {}, acc10 = {}, acc11 = {};

    const unsigned short* arow0 = A + (size_t)(r0 + mL)      * K;
    const unsigned short* arow1 = A + (size_t)(r0 + 16 + mL) * K;

    // Prologue: stage tile 0.
    async_copy_b128(lds_dst0, gsrc);
    wait_asynccnt0();
    __syncthreads();

    int buf = 0;
    for (int kb = 0; kb < K; kb += 32) {
        // Prefetch next weight tile into the other LDS buffer.
        if (kb + 32 < K)
            async_copy_b128(buf ? lds_dst0 : lds_dst1, gsrc + kb + 32);

        // A fragments: contiguous 32B per lane -> 2x global_load_b128 each.
        FragU a0, a1;
#pragma unroll
        for (int v = 0; v < 8; ++v) {
            // A 16x32 f16 layout: VGPR v -> K = 2v + 16h (contiguous, matches B)
            const int ka = 2 * v + 16 * hL;
            a0.u[v] = *(const unsigned int*)(arow0 + kb + ka);
            a1.u[v] = *(const unsigned int*)(arow1 + kb + ka);
        }

        // B fragments from LDS: row (local col), 32B chunk at short offset 16h.
        const unsigned short* sb = smemB[buf];
        FragU b0, b1;
        {
            const uint4* p0 = (const uint4*)(sb + (wn * 32 + mL) * 32 + 16 * hL);
            ((uint4*)b0.u)[0] = p0[0];
            ((uint4*)b0.u)[1] = p0[1];
            const uint4* p1 = (const uint4*)(sb + (wn * 32 + 16 + mL) * 32 + 16 * hL);
            ((uint4*)b1.u)[0] = p1[0];
            ((uint4*)b1.u)[1] = p1[1];
        }

        acc00 = __builtin_amdgcn_wmma_f32_16x16x32_f16(false, a0.h, false, b0.h,
                                                       (short)0, acc00, false, false);
        acc01 = __builtin_amdgcn_wmma_f32_16x16x32_f16(false, a0.h, false, b1.h,
                                                       (short)0, acc01, false, false);
        acc10 = __builtin_amdgcn_wmma_f32_16x16x32_f16(false, a1.h, false, b0.h,
                                                       (short)0, acc10, false, false);
        acc11 = __builtin_amdgcn_wmma_f32_16x16x32_f16(false, a1.h, false, b1.h,
                                                       (short)0, acc11, false, false);

        // Next tile complete + everyone done reading this buffer.
        wait_asynccnt0();
        __syncthreads();
        buf ^= 1;
    }

    v8f accs[2][2] = { { acc00, acc01 }, { acc10, acc11 } };
#pragma unroll
    for (int i = 0; i < 2; ++i)
#pragma unroll
    for (int j = 0; j < 2; ++j) {
        const int coln = c0 + j * 16 + mL;
        const float bval = bias ? bias[coln] : 0.0f;
#pragma unroll
        for (int r = 0; r < 8; ++r) {
            // C/D layout: VGPR r -> M = r (+8 for lanes 16..31); N = lane&15
            const int row = r0 + i * 16 + r + 8 * hL;
            float val = accs[i][j][r] + bval;
            if (flags & EPI_RELU) val = fmaxf(val, 0.0f);
            if (flags & EPI_POS) {
                const int pos = src_pos[row];
                val += pos_table[(size_t)pos * H_DIM + coln];
            }
            const size_t o = (size_t)row * N + coln;
            if (outF32) outF32[o] = val;
            if (outF16) outF16[o] = f2h(val);
        }
    }
}

// ---------------------------------------------------------------------------
// Fused residual-add + LayerNorm. One block (256 thr) per row of 768.
// Writes f32 (next residual / final output) and f16 (next WMMA operand).
// ---------------------------------------------------------------------------
__global__ __launch_bounds__(256) void add_ln_kernel(
    const float* __restrict__ a, const float* __restrict__ res,
    const float* __restrict__ g, const float* __restrict__ beta,
    float* __restrict__ outF32, unsigned short* __restrict__ outF16)
{
    __shared__ float s1[256];
    __shared__ float s2[256];
    const int row = blockIdx.x;
    const int tid = threadIdx.x;
    const float* ap = a   + (size_t)row * H_DIM;
    const float* rp = res + (size_t)row * H_DIM;

    float vals[3];
    float sum = 0.0f, sq = 0.0f;
#pragma unroll
    for (int t = 0; t < 3; ++t) {
        const int i = tid + t * 256;
        const float v = ap[i] + rp[i];
        vals[t] = v; sum += v; sq += v * v;
    }
    s1[tid] = sum; s2[tid] = sq;
    __syncthreads();
    for (int off = 128; off > 0; off >>= 1) {
        if (tid < off) { s1[tid] += s1[tid + off]; s2[tid] += s2[tid + off]; }
        __syncthreads();
    }
    const float mean = s1[0] * (1.0f / H_DIM);
    const float var  = s2[0] * (1.0f / H_DIM) - mean * mean;
    const float inv  = rsqrtf(var + 1e-5f);
#pragma unroll
    for (int t = 0; t < 3; ++t) {
        const int i = tid + t * 256;
        const float y = (vals[t] - mean) * inv * g[i] + beta[i];
        const size_t o = (size_t)row * H_DIM + i;
        if (outF32) outF32[o] = y;
        if (outF16) outF16[o] = f2h(y);
    }
}

// ---------------------------------------------------------------------------
// Sliding-window attention, flash-style online softmax.
// One wave32 per (b, h, s): lane L holds head dims {2L, 2L+1}; each of the
// 65 window positions does a q.k dot via 5-step shfl_xor butterfly.
// Writes ctx directly as f16 (operand of the Wfc WMMA GEMM).
// ---------------------------------------------------------------------------
__global__ __launch_bounds__(256) void swattn_kernel(
    const float* __restrict__ Q, const float* __restrict__ Km,
    const float* __restrict__ V, unsigned short* __restrict__ ctx16)
{
    const int gwave = (blockIdx.x * blockDim.x + threadIdx.x) >> 5;
    const int lane  = threadIdx.x & 31;
    if (gwave >= B_DIM * NH_DIM * S_DIM) return;
    const int s  = gwave % S_DIM;
    const int t2 = gwave / S_DIM;
    const int h  = t2 % NH_DIM;
    const int b  = t2 / NH_DIM;

    const size_t rowbase = ((size_t)(b * S_DIM + s)) * H_DIM + h * HD_DIM;
    const float q0 = Q[rowbase + 2 * lane]     * 0.125f;   // 1/sqrt(64)
    const float q1 = Q[rowbase + 2 * lane + 1] * 0.125f;

    float m = -3.0e38f, l = 0.0f, acc0 = 0.0f, acc1 = 0.0f;
    for (int j = 0; j < 2 * W_WIN + 1; ++j) {
        const int idx = s + j - W_WIN;
        if ((unsigned)idx >= (unsigned)S_DIM) continue;  // masked -> -inf score
        const size_t kb = ((size_t)(b * S_DIM + idx)) * H_DIM + h * HD_DIM;
        float part = q0 * Km[kb + 2 * lane] + q1 * Km[kb + 2 * lane + 1];
#pragma unroll
        for (int off = 16; off > 0; off >>= 1)
            part += __shfl_xor(part, off, 32);
        const float mn    = fmaxf(m, part);
        const float scale = __expf(m - mn);      // 0 on first valid iteration
        const float p     = __expf(part - mn);
        l    = l * scale + p;
        acc0 = acc0 * scale + p * V[kb + 2 * lane];
        acc1 = acc1 * scale + p * V[kb + 2 * lane + 1];
        m = mn;
    }
    const float inv = 1.0f / l;
    ctx16[rowbase + 2 * lane]     = f2h(acc0 * inv);
    ctx16[rowbase + 2 * lane + 1] = f2h(acc1 * inv);
}

// ---------------------------------------------------------------------------
// Host orchestration (stream-ordered; graph-capture safe: no malloc/sync)
// ---------------------------------------------------------------------------
extern "C" void kernel_launch(void* const* d_in, const int* in_sizes, int n_in,
                              void* d_out, int out_size, void* d_ws, size_t ws_size,
                              hipStream_t stream) {
    (void)in_sizes; (void)n_in; (void)out_size; (void)ws_size;

    const float* src_seq   = (const float*)d_in[0];
    const int*   src_pos   = (const int*)  d_in[1];
    const float* pos_table = (const float*)d_in[2];
    const float* W_emb     = (const float*)d_in[3];
    const float* b_emb     = (const float*)d_in[4];
    const float* Wq  = (const float*)d_in[5];  const float* bq  = (const float*)d_in[6];
    const float* Wk  = (const float*)d_in[7];  const float* bk  = (const float*)d_in[8];
    const float* Wv  = (const float*)d_in[9];  const float* bv  = (const float*)d_in[10];
    const float* Wfc = (const float*)d_in[11]; const float* bfc = (const float*)d_in[12];
    const float* ln1g = (const float*)d_in[13]; const float* ln1b = (const float*)d_in[14];
    const float* W1  = (const float*)d_in[15]; const float* b1  = (const float*)d_in[16];
    const float* W2  = (const float*)d_in[17]; const float* b2  = (const float*)d_in[18];
    const float* ln2g = (const float*)d_in[19]; const float* ln2b = (const float*)d_in[20];

    const int M  = MROWS;
    const int nw = H_DIM * H_DIM;

    char* p = (char*)d_ws;
    auto alloc = [&](size_t bytes) -> void* {
        void* r = (void*)p;
        p += (bytes + 255) & ~(size_t)255;
        return r;
    };
    float*          x     = (float*)         alloc((size_t)M * H_DIM * 4);
    unsigned short* x16   = (unsigned short*)alloc((size_t)M * H_DIM * 2);
    float*          q     = (float*)         alloc((size_t)M * H_DIM * 4);
    float*          k     = (float*)         alloc((size_t)M * H_DIM * 4);
    float*          v     = (float*)         alloc((size_t)M * H_DIM * 4);
    float*          tmp   = (float*)         alloc((size_t)M * H_DIM * 4);
    unsigned short* act16 = (unsigned short*)alloc((size_t)M * H_DIM * 2); // padded src / ctx / ffn hidden
    unsigned short* w16   = (unsigned short*)alloc((size_t)nw * 2);        // reusable f16 weight

    const dim3 blk(256);
    const dim3 ggrid(H_DIM / 64, M / 128);   // (12, 32) tiles of 64x128

    // Embedding: x = src_seq @ W_emb.T + b_emb + pos_table[src_pos]
    {
        const int n1 = M * F_PAD;
        cvt_pad_kernel<<<(n1 + 255) / 256, blk, 0, stream>>>(src_seq, act16, M, F_IN, F_PAD);
        const int n2 = H_DIM * F_PAD;
        cvt_pad_kernel<<<(n2 + 255) / 256, blk, 0, stream>>>(W_emb, w16, H_DIM, F_IN, F_PAD);
        gemm_wmma_kernel<<<ggrid, blk, 0, stream>>>(act16, w16, b_emb, x, x16,
                                                    M, H_DIM, F_PAD, EPI_POS,
                                                    pos_table, src_pos);
    }

    const int cvtg = (nw + 255) / 256;
    for (int l = 0; l < L_DIM; ++l) {
        // Q / K / V projections
        cvt_f32_to_f16_kernel<<<cvtg, blk, 0, stream>>>(Wq + (size_t)l * nw, w16, nw);
        gemm_wmma_kernel<<<ggrid, blk, 0, stream>>>(x16, w16, bq + l * H_DIM, q, nullptr,
                                                    M, H_DIM, H_DIM, 0, nullptr, nullptr);
        cvt_f32_to_f16_kernel<<<cvtg, blk, 0, stream>>>(Wk + (size_t)l * nw, w16, nw);
        gemm_wmma_kernel<<<ggrid, blk, 0, stream>>>(x16, w16, bk + l * H_DIM, k, nullptr,
                                                    M, H_DIM, H_DIM, 0, nullptr, nullptr);
        cvt_f32_to_f16_kernel<<<cvtg, blk, 0, stream>>>(Wv + (size_t)l * nw, w16, nw);
        gemm_wmma_kernel<<<ggrid, blk, 0, stream>>>(x16, w16, bv + l * H_DIM, v, nullptr,
                                                    M, H_DIM, H_DIM, 0, nullptr, nullptr);

        // Sliding-window attention -> ctx (f16, feeds Wfc GEMM)
        {
            const int waves = B_DIM * NH_DIM * S_DIM;
            swattn_kernel<<<(waves * 32 + 255) / 256, blk, 0, stream>>>(q, k, v, act16);
        }

        // ctx @ Wfc.T + bfc, then residual + LN1 -> x, x16
        cvt_f32_to_f16_kernel<<<cvtg, blk, 0, stream>>>(Wfc + (size_t)l * nw, w16, nw);
        gemm_wmma_kernel<<<ggrid, blk, 0, stream>>>(act16, w16, bfc + l * H_DIM, tmp, nullptr,
                                                    M, H_DIM, H_DIM, 0, nullptr, nullptr);
        add_ln_kernel<<<M, blk, 0, stream>>>(tmp, x, ln1g + l * H_DIM, ln1b + l * H_DIM, x, x16);

        // FFN: relu(x @ W1.T + b1) @ W2.T + b2, then residual + LN2
        cvt_f32_to_f16_kernel<<<cvtg, blk, 0, stream>>>(W1 + (size_t)l * nw, w16, nw);
        gemm_wmma_kernel<<<ggrid, blk, 0, stream>>>(x16, w16, b1 + l * H_DIM, nullptr, act16,
                                                    M, H_DIM, H_DIM, EPI_RELU, nullptr, nullptr);
        cvt_f32_to_f16_kernel<<<cvtg, blk, 0, stream>>>(W2 + (size_t)l * nw, w16, nw);
        gemm_wmma_kernel<<<ggrid, blk, 0, stream>>>(act16, w16, b2 + l * H_DIM, tmp, nullptr,
                                                    M, H_DIM, H_DIM, 0, nullptr, nullptr);

        float* outp = (l == L_DIM - 1) ? (float*)d_out : x;
        add_ln_kernel<<<M, blk, 0, stream>>>(tmp, x, ln2g + l * H_DIM, ln2b + l * H_DIM, outp, x16);
    }
}
